// HGATLayer_6451040878625
// MI455X (gfx1250) — matrix operand order repaired and checked by reading, not compiled, not run on hardware
//
#include <hip/hip_runtime.h>
#include <hip/hip_bf16.h>
#include <math.h>

// ---------------- problem constants ----------------
#define HEADS   4
#define CDIM    128            // per-head channels
#define DDIM    128            // input feature dim
#define HC      512            // HEADS*CDIM
#define NEG_SLOPE 0.2f
#define LN_EPS    1e-5f
#define DROP_THRESH 858993459u // 0.2 * 2^32  (keep if hash >= thresh)
#define DROP_SCALE  1.25f      // 1/(1-0.2)

typedef __attribute__((ext_vector_type(16))) _Float16 v16h;
typedef __attribute__((ext_vector_type(8)))  _Float16 v8h;
typedef __attribute__((ext_vector_type(8)))  float    v8f;

// ---------------- small helpers ----------------
__device__ __forceinline__ unsigned hash_u32(unsigned x) {
  x ^= x >> 16; x *= 0x7feb352dU;
  x ^= x >> 15; x *= 0x846ca68bU;
  x ^= x >> 16;
  return x;
}

__device__ __forceinline__ void atomicMaxF(float* addr, float val) {
  int old = __float_as_int(*addr);
  while (__int_as_float(old) < val) {
    int assumed = old;
    old = atomicCAS((int*)addr, assumed, __float_as_int(val));
    if (old == assumed) break;
  }
}

__device__ __forceinline__ float waveReduceSum(float v) {
  #pragma unroll
  for (int off = 16; off > 0; off >>= 1) v += __shfl_xor(v, off, 32);
  return v;
}

// ---------------- conversion / transpose ----------------
__global__ void k_cvt_f16(const float* __restrict__ src, _Float16* __restrict__ dst, int n) {
  int i = blockIdx.x * blockDim.x + threadIdx.x;
  int stride = gridDim.x * blockDim.x;
  for (; i < n; i += stride) dst[i] = (_Float16)src[i];
}

// W: [DDIM, HC] row-major fp32  ->  WT: [HC, DDIM] row-major f16
__global__ void k_transpose_w(const float* __restrict__ W, _Float16* __restrict__ WT) {
  int i = blockIdx.x * blockDim.x + threadIdx.x;
  int stride = gridDim.x * blockDim.x;
  const int total = HC * DDIM;
  for (; i < total; i += stride) {
    int n = i / DDIM;
    int k = i - n * DDIM;
    WT[i] = (_Float16)W[k * HC + n];
  }
}

// ---------------- WMMA GEMM:  Y[N,512] = A[N,128](f16) * W[128,512] + bias ----------------
// BT is the transposed weight [512,128] f16 so B fragments are K-contiguous per lane.
// One wave computes a 16x64 output strip (4 col-tiles): each A fragment is loaded once
// per K-step and reused across 4 B fragments -> 16 x v_wmma_f32_16x16x32_f16 per wave,
// 4 back-to-back WMMA issues per A-load for better XDL latency hiding.
__global__ void __launch_bounds__(256) k_gemm_wmma(
    const _Float16* __restrict__ A, const _Float16* __restrict__ BT,
    const float* __restrict__ bias, float* __restrict__ Y, int nRows) {
  const int lane = threadIdx.x & 31;
  const int wave = threadIdx.x >> 5;
  const int tile = blockIdx.x * 8 + wave;
  const int colGroups = HC / 64;                // 8 groups of 4 col-tiles
  const int tr = tile / colGroups;
  const int tg = tile - tr * colGroups;
  if (tr * 16 >= nRows) return;                 // wave-uniform; never taken for exact launch

  const int hl = lane & 15;                     // row (A) / col (B,C) within tile
  const int hi = lane >> 4;                     // half-wave select

  const _Float16* __restrict__ arow = A + (size_t)(tr * 16 + hl) * DDIM;
  // B fragment row (= output column) for col-tile t: tg*64 + t*16 + hl
  const _Float16* __restrict__ bbase = BT + (size_t)(tg * 64 + hl) * DDIM;

  v8f c0 = {}, c1 = {}, c2 = {}, c3 = {};
  #pragma unroll
  for (int kk = 0; kk < DDIM; kk += 32) {
    // A 16x32 f16 fragment: lanes 0-15 hold K kk..kk+15, lanes 16-31 hold K kk+16..kk+31.
    v16h a = *(const v16h*)(arow + kk + hi * 16);
    // B 32x16 f16 fragments: lanes 0-15 K kk..kk+15, lanes 16-31 K kk+16..kk+31.
    const size_t bo = (size_t)kk + hi * 16;
    v16h b0 = *(const v16h*)(bbase + 0 * 16 * DDIM + bo);
    v16h b1 = *(const v16h*)(bbase + 1 * 16 * DDIM + bo);
    v16h b2 = *(const v16h*)(bbase + 2 * 16 * DDIM + bo);
    v16h b3 = *(const v16h*)(bbase + 3 * 16 * DDIM + bo);
    c0 = __builtin_amdgcn_wmma_f32_16x16x32_f16(false, a, false, b0, (short)0, c0, false, false);
    c1 = __builtin_amdgcn_wmma_f32_16x16x32_f16(false, a, false, b1, (short)0, c1, false, false);
    c2 = __builtin_amdgcn_wmma_f32_16x16x32_f16(false, a, false, b2, (short)0, c2, false, false);
    c3 = __builtin_amdgcn_wmma_f32_16x16x32_f16(false, a, false, b3, (short)0, c3, false, false);
  }

  // D layout: VGPR r -> M = r + 8*hi, N = hl
  #pragma unroll
  for (int t = 0; t < 4; ++t) {
    const v8f& c = (t == 0) ? c0 : (t == 1) ? c1 : (t == 2) ? c2 : c3;
    const int col = tg * 64 + t * 16 + hl;
    const float bv = bias[col];
    #pragma unroll
    for (int r = 0; r < 8; ++r) {
      int row = tr * 16 + r + hi * 8;
      Y[(size_t)row * HC + col] = c[r] + bv;
    }
  }
}

// ---------------- segment softmax pipeline ----------------
__global__ void k_init_md(float* __restrict__ m, float* __restrict__ den,
                          float* __restrict__ acc, int nDst) {
  int i = blockIdx.x * blockDim.x + threadIdx.x;
  int stride = gridDim.x * blockDim.x;
  const int total = nDst * HC;
  for (; i < total; i += stride) {
    acc[i] = 0.f;
    if (i < nDst * HEADS) { m[i] = -3.0e38f; den[i] = 0.f; }
  }
}

// One wave per edge: score[e][h] = sum_c att[h][c]*leakyrelu(xl[src]+xr[dst]); seg-max into m.
__global__ void k_edge_scores(const float* __restrict__ xl, const float* __restrict__ xr,
                              const int* __restrict__ srcI, const int* __restrict__ dstI,
                              const float* __restrict__ att,
                              float* __restrict__ score, float* __restrict__ m,
                              int nE, int nLoops) {
  const int wid  = blockIdx.x * (blockDim.x >> 5) + (threadIdx.x >> 5);
  const int lane = threadIdx.x & 31;
  const int total = nE + nLoops;
  if (wid >= total) return;
  int s, d;
  if (wid < nE) { s = srcI[wid]; d = dstI[wid]; } else { s = d = wid - nE; }
  const float* pl = xl + (size_t)s * HC;
  const float* pr = xr + (size_t)d * HC;
  #pragma unroll
  for (int h = 0; h < HEADS; ++h) {
    float acc = 0.f;
    #pragma unroll
    for (int j = 0; j < 4; ++j) {
      int c = lane + j * 32;
      float v = pl[h * CDIM + c] + pr[h * CDIM + c];
      v = v > 0.f ? v : v * NEG_SLOPE;
      acc += att[h * CDIM + c] * v;
    }
    acc = waveReduceSum(acc);
    if (lane == 0) {
      score[(size_t)wid * HEADS + h] = acc;
      atomicMaxF(&m[d * HEADS + h], acc);
    }
  }
}

// One thread per (edge, head): score := exp(score - m[dst]); seg-sum into den.
__global__ void k_edge_expsum(float* __restrict__ score, const float* __restrict__ m,
                              float* __restrict__ den, const int* __restrict__ dstI,
                              int nE, int nLoops) {
  int idx = blockIdx.x * blockDim.x + threadIdx.x;
  const int total = (nE + nLoops) * HEADS;
  if (idx >= total) return;
  int e = idx >> 2, h = idx & 3;
  int d = (e < nE) ? dstI[e] : (e - nE);
  float ex = __expf(score[idx] - m[d * HEADS + h]);
  score[idx] = ex;
  atomicAdd(&den[d * HEADS + h], ex);
}

// One wave per edge: acc[dst] += (ex/den[dst]) * xl[src]
__global__ void k_edge_scatter(const float* __restrict__ xl, const float* __restrict__ score,
                               const float* __restrict__ den,
                               const int* __restrict__ srcI, const int* __restrict__ dstI,
                               float* __restrict__ acc, int nE, int nLoops) {
  const int wid  = blockIdx.x * (blockDim.x >> 5) + (threadIdx.x >> 5);
  const int lane = threadIdx.x & 31;
  const int total = nE + nLoops;
  if (wid >= total) return;
  int s, d;
  if (wid < nE) { s = srcI[wid]; d = dstI[wid]; } else { s = d = wid - nE; }
  const float* pl = xl + (size_t)s * HC;
  float* pd = acc + (size_t)d * HC;
  #pragma unroll
  for (int h = 0; h < HEADS; ++h) {
    float alpha = score[(size_t)wid * HEADS + h] / den[d * HEADS + h];
    #pragma unroll
    for (int j = 0; j < 4; ++j) {
      int c = h * CDIM + lane + j * 32;
      atomicAdd(&pd[c], alpha * pl[c]);
    }
  }
}

// out[n,c] = mean_h acc[n,h,c] + bias[c]
__global__ void k_head_mean(const float* __restrict__ acc, const float* __restrict__ bias,
                            float* __restrict__ out, int nDst) {
  int i = blockIdx.x * blockDim.x + threadIdx.x;
  int stride = gridDim.x * blockDim.x;
  const int total = nDst * CDIM;
  for (; i < total; i += stride) {
    int n = i >> 7, c = i & 127;
    const float* p = acc + (size_t)n * HC + c;
    out[i] = 0.25f * (p[0] + p[CDIM] + p[2 * CDIM] + p[3 * CDIM]) + bias[c];
  }
}

// y = dropout(layernorm(elu(scale*(a [+ b])) + resid))   — one wave per row
__global__ void k_finalize(const float* __restrict__ a, const float* __restrict__ b,
                           float scaleAB, const float* __restrict__ resid,
                           const float* __restrict__ gamma, const float* __restrict__ beta,
                           float* __restrict__ y, int nRows, unsigned seed) {
  const int wid  = blockIdx.x * (blockDim.x >> 5) + (threadIdx.x >> 5);
  const int lane = threadIdx.x & 31;
  if (wid >= nRows) return;
  float v[4];
  float sum = 0.f;
  #pragma unroll
  for (int j = 0; j < 4; ++j) {
    int c = lane + j * 32;
    size_t idx = (size_t)wid * CDIM + c;
    float t = a[idx];
    if (b) t += b[idx];
    t *= scaleAB;
    t = t > 0.f ? t : (__expf(t) - 1.f);   // ELU
    t += resid[idx];
    v[j] = t;
    sum += t;
  }
  sum = waveReduceSum(sum);
  const float mu = sum * (1.f / CDIM);
  float var = 0.f;
  #pragma unroll
  for (int j = 0; j < 4; ++j) { float d = v[j] - mu; var += d * d; }
  var = waveReduceSum(var) * (1.f / CDIM);
  const float rstd = rsqrtf(var + LN_EPS);
  #pragma unroll
  for (int j = 0; j < 4; ++j) {
    int c = lane + j * 32;
    size_t idx = (size_t)wid * CDIM + c;
    float o = (v[j] - mu) * rstd * gamma[c] + beta[c];
    unsigned hsh = hash_u32(seed * 0x9E3779B9u + (unsigned)idx);
    y[idx] = (hsh >= DROP_THRESH) ? o * DROP_SCALE : 0.f;
  }
}

// ---------------- host orchestration ----------------
extern "C" void kernel_launch(void* const* d_in, const int* in_sizes, int n_in,
                              void* d_out, int out_size, void* d_ws, size_t ws_size,
                              hipStream_t stream) {
  const float* x_met = (const float*)d_in[0];
  const float* x_rxn = (const float*)d_in[1];
  const int* ei_sub    = (const int*)d_in[2];
  const int* ei_prod   = (const int*)d_in[3];
  const int* ei_shared = (const int*)d_in[4];

  const int nMet = in_sizes[0] / DDIM;
  const int nRxn = in_sizes[1] / DDIM;
  const int E_sub    = in_sizes[2] / 2;
  const int E_prod   = in_sizes[3] / 2;
  const int E_shared = in_sizes[4] / 2;
  const int nMax = nMet > nRxn ? nMet : nRxn;

  // weight pointers per dict order: Wl, Wr, att, bl, br, bias for (sub, prod, shared)
  const float* Wl_sub = (const float*)d_in[5];   const float* Wr_sub = (const float*)d_in[6];
  const float* att_sub = (const float*)d_in[7];  const float* bl_sub = (const float*)d_in[8];
  const float* br_sub = (const float*)d_in[9];   const float* bias_sub = (const float*)d_in[10];
  const float* Wl_prod = (const float*)d_in[11]; const float* Wr_prod = (const float*)d_in[12];
  const float* att_prod = (const float*)d_in[13];const float* bl_prod = (const float*)d_in[14];
  const float* br_prod = (const float*)d_in[15]; const float* bias_prod = (const float*)d_in[16];
  const float* Wl_sh = (const float*)d_in[17];   const float* Wr_sh = (const float*)d_in[18];
  const float* att_sh = (const float*)d_in[19];  const float* bl_sh = (const float*)d_in[20];
  const float* br_sh = (const float*)d_in[21];   const float* bias_sh = (const float*)d_in[22];
  const float* gamma_rxn = (const float*)d_in[23]; const float* beta_rxn = (const float*)d_in[24];
  const float* gamma_met = (const float*)d_in[25]; const float* beta_met = (const float*)d_in[26];

  // ---- workspace carve-out ----
  char* base = (char*)d_ws;
  size_t off = 0;
  auto carve = [&](size_t bytes) -> char* {
    char* p = base + off;
    off = (off + bytes + 255) & ~(size_t)255;
    return p;
  };
  _Float16* xh_met = (_Float16*)carve((size_t)nMet * DDIM * 2);
  _Float16* xh_rxn = (_Float16*)carve((size_t)nRxn * DDIM * 2);
  _Float16* wlT    = (_Float16*)carve((size_t)HC * DDIM * 2);
  _Float16* wrT    = (_Float16*)carve((size_t)HC * DDIM * 2);
  float* xl    = (float*)carve((size_t)nMax * HC * 4);
  float* xr    = (float*)carve((size_t)nMax * HC * 4);
  float* score = (float*)carve((size_t)(E_shared + nRxn) * HEADS * 4); // max edge count
  float* mbuf  = (float*)carve((size_t)nMax * HEADS * 4);
  float* dbuf  = (float*)carve((size_t)nMax * HEADS * 4);
  float* accb  = (float*)carve((size_t)nMax * HC * 4);
  float* out_sub    = (float*)carve((size_t)nRxn * CDIM * 4);
  float* out_shared = (float*)carve((size_t)nRxn * CDIM * 4);
  float* out_prod   = (float*)carve((size_t)nMet * CDIM * 4);
  (void)ws_size;

  const int TB = 256;
  auto gsBlocks = [](int n, int tb) { int b = (n + tb - 1) / tb; return b > 4096 ? 4096 : b; };

  // ---- f16 feature copies ----
  k_cvt_f16<<<gsBlocks(nMet * DDIM, TB), TB, 0, stream>>>(x_met, xh_met, nMet * DDIM);
  k_cvt_f16<<<gsBlocks(nRxn * DDIM, TB), TB, 0, stream>>>(x_rxn, xh_rxn, nRxn * DDIM);

  // ---- one relation pass ----
  auto run_relation = [&](const _Float16* xsH, int nSrc, const _Float16* xdH, int nDst,
                          const int* ei, int nE, int nLoops,
                          const float* Wl, const float* bl, const float* Wr, const float* br,
                          const float* att, const float* bias, float* outBuf) {
    const int* srcI = ei;
    const int* dstI = ei + nE;
    // weight transposes to f16
    k_transpose_w<<<gsBlocks(HC * DDIM, TB), TB, 0, stream>>>(Wl, wlT);
    k_transpose_w<<<gsBlocks(HC * DDIM, TB), TB, 0, stream>>>(Wr, wrT);
    // GEMMs (WMMA): xl = xs @ Wl + bl ; xr = xd @ Wr + br
    {
      int tilesL = (nSrc / 16) * (HC / 64);   // one 16x64 strip per wave
      int tilesR = (nDst / 16) * (HC / 64);
      k_gemm_wmma<<<(tilesL + 7) / 8, TB, 0, stream>>>(xsH, wlT, bl, xl, nSrc);
      k_gemm_wmma<<<(tilesR + 7) / 8, TB, 0, stream>>>(xdH, wrT, br, xr, nDst);
    }
    // segment softmax + aggregate
    k_init_md<<<gsBlocks(nDst * HC, TB), TB, 0, stream>>>(mbuf, dbuf, accb, nDst);
    int totalE = nE + nLoops;
    int edgeBlocks = (totalE + 7) / 8;   // 8 waves per block, 1 edge per wave
    k_edge_scores<<<edgeBlocks, TB, 0, stream>>>(xl, xr, srcI, dstI, att, score, mbuf, nE, nLoops);
    k_edge_expsum<<<(totalE * HEADS + TB - 1) / TB, TB, 0, stream>>>(score, mbuf, dbuf, dstI, nE, nLoops);
    k_edge_scatter<<<edgeBlocks, TB, 0, stream>>>(xl, score, dbuf, srcI, dstI, accb, nE, nLoops);
    k_head_mean<<<gsBlocks(nDst * CDIM, TB), TB, 0, stream>>>(accb, bias, outBuf, nDst);
  };

  // sub:    met -> rxn, no self loops
  run_relation(xh_met, nMet, xh_rxn, nRxn, ei_sub, E_sub, 0,
               Wl_sub, bl_sub, Wr_sub, br_sub, att_sub, bias_sub, out_sub);
  // shared: rxn -> rxn, + self loops
  run_relation(xh_rxn, nRxn, xh_rxn, nRxn, ei_shared, E_shared, nRxn,
               Wl_sh, bl_sh, Wr_sh, br_sh, att_sh, bias_sh, out_shared);
  // prod:   rxn -> met, no self loops
  run_relation(xh_rxn, nRxn, xh_met, nMet, ei_prod, E_prod, 0,
               Wl_prod, bl_prod, Wr_prod, br_prod, att_prod, bias_prod, out_prod);

  // ---- epilogue: mean-combine, ELU, residual, LayerNorm, MC-dropout ----
  float* y_rxn = (float*)d_out;
  float* y_met = (float*)d_out + (size_t)nRxn * CDIM;
  k_finalize<<<(nRxn + 7) / 8, TB, 0, stream>>>(out_sub, out_shared, 0.5f, x_rxn,
                                                gamma_rxn, beta_rxn, y_rxn, nRxn, 101u);
  k_finalize<<<(nMet + 7) / 8, TB, 0, stream>>>(out_prod, (const float*)nullptr, 1.0f, x_met,
                                                gamma_met, beta_met, y_met, nMet, 202u);
  (void)n_in; (void)out_size;
}
